// EnhancedPGATActor_57286273794131
// MI455X (gfx1250) — matrix operand order repaired and checked by reading, not compiled, not run on hardware
//
#include <hip/hip_runtime.h>

// ---------------------------------------------------------------------------
// EnhancedPGATActor: 2-layer positional GAT (H=4, C=128, HC=512) + MLP head.
// Dense GEMMs: CDNA5 WMMA (v_wmma_f32_16x16x32_bf16, f32 accumulate).
//  - each wave computes a 16x32 output strip (A fragment reused by 2 WMMAs)
//  - block's 32-column B tile staged once in LDS, shared by all 4 waves
//  - software-pipelined inner loop (next fragments fetched before WMMA)
//  - unconditional speculative global_prefetch on the A stream
//  - bf16 outputs fused into the epilogue (no separate conversion pass)
// Edge softmax/aggregation: gather/scatter through L2 (20MB node tensors fit
// the 192MB L2), VALU + f32 atomics.
// ---------------------------------------------------------------------------

typedef __attribute__((ext_vector_type(16))) __bf16 v16bf;
typedef __attribute__((ext_vector_type(8)))  __bf16 v8bf;
typedef __attribute__((ext_vector_type(8)))  float  v8f;

static constexpr int N_NODES = 10000;
static constexpr int N_EDGES = 160000;
static constexpr int Hh      = 4;
static constexpr int Cc      = 128;
static constexpr int HC      = 512;   // H*C
static constexpr int HID     = 128;
static constexpr int MPAD    = 10048; // 157 * 64, multiple of the 64-row block tile
static constexpr int KMAX    = 512;   // largest GEMM K (for LDS B tile)
static constexpr float SCALE = 0.17677669529663687f; // (HID/H)^-0.5 = 32^-0.5

// ------------------------------ helpers ------------------------------------

__device__ __forceinline__ __bf16 f2bf(float x) {
    unsigned u = __float_as_uint(x);
    unsigned r = (u + 0x7FFFu + ((u >> 16) & 1u)) >> 16;   // round-to-nearest-even
    return __builtin_bit_cast(__bf16, (unsigned short)r);
}

__device__ __forceinline__ void atomicMaxF(float* addr, float val) {
    unsigned* a = (unsigned*)addr;
    unsigned old = *a;
    while (__uint_as_float(old) < val) {
        unsigned assumed = old;
        old = atomicCAS(a, assumed, __float_as_uint(val));
        if (old == assumed) break;
    }
}

// ------------------------------ fills --------------------------------------

__global__ void fill_f32(float* p, float v, long long n) {
    long long i = (long long)blockIdx.x * blockDim.x + threadIdx.x;
    if (i < n) p[i] = v;
}

__global__ void fill_bf16_zero(__bf16* p, long long n) {
    long long i = (long long)blockIdx.x * blockDim.x + threadIdx.x;
    if (i < n) p[i] = __builtin_bit_cast(__bf16, (unsigned short)0);
}

// ----------------------- layer-0 feature packing ---------------------------
// qf = obs[:, 0:12] ++ obs[:, 30:37]   (19 -> pad 32)
// kf = obs[:, 12:24] ++ obs[:, 26:30]  (16 -> pad 32)
// vf = obs[:, 12:30]                   (18 -> pad 32)
__global__ void pack_qkv0(const float* __restrict__ obs,
                          __bf16* __restrict__ Aq, __bf16* __restrict__ Ak,
                          __bf16* __restrict__ Av) {
    long long i = (long long)blockIdx.x * blockDim.x + threadIdx.x;
    if (i >= (long long)MPAD * 32) return;
    int r = (int)(i >> 5), c = (int)(i & 31);
    float q = 0.f, k = 0.f, v = 0.f;
    if (r < N_NODES) {
        const float* o = obs + (long long)r * 37;
        if (c < 12) q = o[c]; else if (c < 19) q = o[30 + (c - 12)];
        if (c < 12) k = o[12 + c]; else if (c < 16) k = o[26 + (c - 12)];
        if (c < 18) v = o[12 + c];
    }
    Aq[i] = f2bf(q); Ak[i] = f2bf(k); Av[i] = f2bf(v);
}

// W (K x N, f32, row-major) -> Bt (N x Kpad, bf16), zero-padded in K.
__global__ void pack_w_t(const float* __restrict__ W, __bf16* __restrict__ Bt,
                         int K, int Kpad, int N) {
    long long i = (long long)blockIdx.x * blockDim.x + threadIdx.x;
    if (i >= (long long)N * Kpad) return;
    int n = (int)(i / Kpad), kk = (int)(i % Kpad);
    Bt[i] = f2bf(kk < K ? W[(long long)kk * N + n] : 0.f);
}

// X (M x N, f32) -> Y (MPAD x N, bf16), rows >= M zero-padded.
__global__ void cvt_pad_bf16(const float* __restrict__ X, __bf16* __restrict__ Y,
                             int M, int N) {
    long long i = (long long)blockIdx.x * blockDim.x + threadIdx.x;
    if (i >= (long long)MPAD * N) return;
    int r = (int)(i / N);
    Y[i] = f2bf(r < M ? X[i] : 0.f);
}

// ----------------------------- WMMA GEMM -----------------------------------
// C[M,N] = A[MPAD,K](bf16) * Bt[N,K](bf16, transposed) (+bias) (relu?)
// Block = 128 threads = 4 waves; each wave computes a 16x32 output strip
// (two 16x16 WMMA tiles sharing one A fragment). The block's 32-column B
// tile is staged once in LDS and shared by all 4 waves. Inner loop is
// software-pipelined: next-step fragments are loaded before current WMMAs.
// Fragment layouts as loaded by this kernel (wave32; contiguous k, not the ISA table order):
//   A 16x32 bf16 : lane l holds row l&15; lanes<16 hold K 0..15,
//                  lanes>=16 hold K 16..31.
//   B 32x16 bf16 : lane l holds col l&15; lanes<16 hold K 0..15,
//                  lanes>=16 hold K 16..31 (contiguous in transposed layout).
//   C/D 16x16 f32: lane l holds col l&15; vgpr i -> row i (lanes<16) / 8+i.
__global__ __launch_bounds__(128)
void gemm_bf16_wmma(const __bf16* __restrict__ A, const __bf16* __restrict__ Bt,
                    const float* __restrict__ bias,
                    float* __restrict__ C, __bf16* __restrict__ Cbf,
                    int M, int N, int K, int relu) {
    __shared__ __align__(64) __bf16 sB[32 * KMAX];

    const int lane    = threadIdx.x & 31;
    const int wave    = threadIdx.x >> 5;
    const int l15     = lane & 15;
    const int half    = lane >> 4;              // which half-wave
    const int row0    = blockIdx.x * 64 + wave * 16;
    const int colBase = blockIdx.y * 32;

    // Cooperative stage of the B tile: 32 cols x K, contiguous per column.
    {
        const v8bf* gsrc = (const v8bf*)(Bt + (long long)colBase * K);
        v8bf* ldst = (v8bf*)sB;
        const int n8 = 4 * K;                   // (32*K)/8 v8bf chunks
        for (int t = threadIdx.x; t < n8; t += 128) ldst[t] = gsrc[t];
    }
    __syncthreads();

    const __bf16* arow  = A + (long long)(row0 + l15) * K + half * 16;
    const __bf16* bcol0 = sB + (long long)l15 * K + half * 16;
    const __bf16* bcol1 = bcol0 + (long long)16 * K;

    v8f acc0 = {}, acc1 = {};
    const int steps = K >> 5;

    // prologue fetch (step 0)
    v8bf  alo = *(const v8bf*)(arow);
    v8bf  ahi = *(const v8bf*)(arow + 8);
    v16bf bv0 = *(const v16bf*)(bcol0);
    v16bf bv1 = *(const v16bf*)(bcol1);

    int k0 = 0;
    for (int s = 0; s < steps; ++s) {
        const int kn = k0 + ((s + 1 < steps) ? 32 : 0);   // last step refetches self
        __builtin_prefetch((const void*)(arow + kn + 32), 0, 0);  // speculative
        v8bf  alo_n = *(const v8bf*)(arow + kn);
        v8bf  ahi_n = *(const v8bf*)(arow + kn + 8);
        v16bf bv0_n = *(const v16bf*)(bcol0 + kn);
        v16bf bv1_n = *(const v16bf*)(bcol1 + kn);

        v16bf av;
        #pragma unroll
        for (int i = 0; i < 8; ++i) { av[i] = alo[i]; av[i + 8] = ahi[i]; }

        acc0 = __builtin_amdgcn_wmma_f32_16x16x32_bf16(
            false, av, false, bv0, (short)0, acc0, false, false);
        acc1 = __builtin_amdgcn_wmma_f32_16x16x32_bf16(
            false, av, false, bv1, (short)0, acc1, false, false);

        alo = alo_n; ahi = ahi_n; bv0 = bv0_n; bv1 = bv1_n;
        k0 = kn;
    }

    const int col0 = colBase + l15;
    const int col1 = col0 + 16;
    const float b0 = bias ? bias[col0] : 0.0f;
    const float b1 = bias ? bias[col1] : 0.0f;
    #pragma unroll
    for (int i = 0; i < 8; ++i) {
        int m = row0 + (half ? 8 + i : i);
        if (m < M) {
            float v0 = acc0[i] + b0;
            float v1 = acc1[i] + b1;
            if (relu) { v0 = fmaxf(v0, 0.0f); v1 = fmaxf(v1, 0.0f); }
            if (C) {
                C[(long long)m * N + col0] = v0;
                C[(long long)m * N + col1] = v1;
            }
            if (Cbf) {
                Cbf[(long long)m * N + col0] = f2bf(v0);
                Cbf[(long long)m * N + col1] = f2bf(v1);
            }
        }
    }
}

// ------------------------------ edge pass 1 --------------------------------
// One wave per (edge, head): alpha = SCALE * exp(-|pos_d - pos_s|) * <q_d, k_s>
__global__ void edge_alpha_max(const float* __restrict__ q, const float* __restrict__ k,
                               const float* __restrict__ obs,
                               const int* __restrict__ src, const int* __restrict__ dst,
                               float* __restrict__ alphaRaw, float* __restrict__ amax) {
    int gw   = (int)(((long long)blockIdx.x * blockDim.x + threadIdx.x) >> 5);
    int lane = threadIdx.x & 31;
    if (gw >= N_EDGES * Hh) return;        // uniform across the wave
    int e = gw >> 2, h = gw & 3;
    int s = src[e], d = dst[e];

    float dx = obs[(long long)d * 37 + 0] - obs[(long long)s * 37 + 0];
    float dy = obs[(long long)d * 37 + 1] - obs[(long long)s * 37 + 1];
    float pw = __expf(-sqrtf(dx * dx + dy * dy));

    const float* qp = q + (long long)d * HC + h * Cc;
    const float* kp = k + (long long)s * HC + h * Cc;
    float acc = 0.f;
    #pragma unroll
    for (int i = 0; i < Cc / 32; ++i) acc += qp[lane + 32 * i] * kp[lane + 32 * i];
    #pragma unroll
    for (int off = 16; off; off >>= 1) acc += __shfl_xor(acc, off, 32);

    if (lane == 0) {
        float a = acc * SCALE * pw;
        alphaRaw[e * Hh + h] = a;
        atomicMaxF(&amax[d * Hh + h], a);
    }
}

// ------------------------------ edge pass 2 --------------------------------
__global__ void edge_exp_sum(const int* __restrict__ dst,
                             float* __restrict__ alphaRaw,
                             const float* __restrict__ amax,
                             float* __restrict__ den) {
    int i = blockIdx.x * blockDim.x + threadIdx.x;
    if (i >= N_EDGES * Hh) return;
    int e = i >> 2, h = i & 3;
    int d = dst[e];
    float ex = __expf(alphaRaw[i] - amax[d * Hh + h]);
    alphaRaw[i] = ex;
    atomicAdd(&den[d * Hh + h], ex);
}

// ------------------------------ edge pass 3 --------------------------------
// agg[dst, :] += v[src, :] * ex[e,h]/(den[dst,h]+eps). Block (128 thr) per edge.
__global__ __launch_bounds__(128)
void edge_aggregate(const float* __restrict__ v,
                    const int* __restrict__ src, const int* __restrict__ dst,
                    const float* __restrict__ exw, const float* __restrict__ den,
                    float* __restrict__ agg) {
    int e = blockIdx.x;
    int s = src[e], d = dst[e];
    for (int c = threadIdx.x; c < HC; c += 128) {
        int h = c >> 7;
        float w = exw[e * Hh + h] / (den[d * Hh + h] + 1e-16f);
        atomicAdd(&agg[(long long)d * HC + c], v[(long long)s * HC + c] * w);
    }
}

// --------------------------- final 256->2 head ------------------------------
__global__ void final_head(const float* __restrict__ x, const float* __restrict__ w3,
                           const float* __restrict__ b3, float* __restrict__ out) {
    int i = blockIdx.x * blockDim.x + threadIdx.x;
    if (i >= N_NODES * 2) return;
    int r = i >> 1, o = i & 1;
    const float* xr = x + (long long)r * 256;
    float acc = b3[o];
    #pragma unroll 8
    for (int j = 0; j < 256; ++j) acc += xr[j] * w3[j * 2 + o];
    out[i] = acc;
}

// ---------------------------------------------------------------------------

static inline dim3 g1(long long n, int b) { return dim3((unsigned)((n + b - 1) / b)); }

extern "C" void kernel_launch(void* const* d_in, const int* in_sizes, int n_in,
                              void* d_out, int out_size, void* d_ws, size_t ws_size,
                              hipStream_t stream) {
    const float* obs  = (const float*)d_in[0];
    const int*   eidx = (const int*)  d_in[1];
    const float* wq0  = (const float*)d_in[2];
    const float* wk0  = (const float*)d_in[3];
    const float* wv0  = (const float*)d_in[4];
    const float* wo0  = (const float*)d_in[5];
    const float* bo0  = (const float*)d_in[6];
    const float* wq1  = (const float*)d_in[7];
    const float* wk1  = (const float*)d_in[8];
    const float* wv1  = (const float*)d_in[9];
    const float* wo1  = (const float*)d_in[10];
    const float* bo1  = (const float*)d_in[11];
    const float* w1   = (const float*)d_in[12];
    const float* b1   = (const float*)d_in[13];
    const float* w2   = (const float*)d_in[14];
    const float* b2   = (const float*)d_in[15];
    const float* w3   = (const float*)d_in[16];
    const float* b3   = (const float*)d_in[17];
    float* out = (float*)d_out;

    const int* src = eidx;
    const int* dst = eidx + N_EDGES;

    // ---- workspace bump allocator ----
    char* ws = (char*)d_ws;
    size_t off = 0;
    auto alloc = [&](size_t bytes) -> void* {
        off = (off + 255) & ~(size_t)255;
        void* p = ws + off;
        off += bytes;
        return p;
    };
    __bf16* Aq0    = (__bf16*)alloc((size_t)MPAD * 32 * 2);
    __bf16* Ak0    = (__bf16*)alloc((size_t)MPAD * 32 * 2);
    __bf16* Av0    = (__bf16*)alloc((size_t)MPAD * 32 * 2);
    __bf16* Bt     = (__bf16*)alloc((size_t)512 * 512 * 2);   // reused per GEMM
    float*  fq     = (float*) alloc((size_t)N_NODES * HC * 4);
    float*  fk     = (float*) alloc((size_t)N_NODES * HC * 4);
    float*  fv     = (float*) alloc((size_t)N_NODES * HC * 4);
    float*  amax   = (float*) alloc((size_t)N_NODES * Hh * 4);
    float*  den    = (float*) alloc((size_t)N_NODES * Hh * 4);
    float*  exw    = (float*) alloc((size_t)N_EDGES * Hh * 4);
    float*  agg    = (float*) alloc((size_t)N_NODES * HC * 4);
    __bf16* bfA512 = (__bf16*)alloc((size_t)MPAD * 512 * 2);  // agg stage / Mx256 stage
    __bf16* bfA128 = (__bf16*)alloc((size_t)MPAD * 128 * 2);  // h stage
    float*  pad    = (float*) alloc((size_t)4096);            // guard after last A buffer
    (void)pad;
    float*  m2     = fk;           // lifetime disjoint with layer-1 k
    __bf16* b256   = bfA512;       // Mx256 bf16 stage aliases the 512 buffer
    (void)ws_size; (void)in_sizes; (void)n_in; (void)out_size;

    const int TB = 256;
    auto gemm = [&](const __bf16* A, const float* W, int K, int Kpad, int N,
                    const float* bias, float* Cf, __bf16* Cb, int relu) {
        pack_w_t<<<g1((long long)N * Kpad, TB), TB, 0, stream>>>(W, Bt, K, Kpad, N);
        dim3 grid(MPAD / 64, N / 32);
        gemm_bf16_wmma<<<grid, 128, 0, stream>>>(A, Bt, bias, Cf, Cb, N_NODES, N, Kpad, relu);
    };
    auto edge_layer = [&](const float* q, const float* k, const float* v) {
        fill_f32<<<g1(N_NODES * Hh, TB), TB, 0, stream>>>(amax, -3.0e38f, N_NODES * Hh);
        fill_f32<<<g1(N_NODES * Hh, TB), TB, 0, stream>>>(den, 0.f, N_NODES * Hh);
        fill_f32<<<g1((long long)N_NODES * HC, TB), TB, 0, stream>>>(agg, 0.f, (long long)N_NODES * HC);
        edge_alpha_max<<<g1((long long)N_EDGES * Hh * 32, TB), TB, 0, stream>>>(
            q, k, obs, src, dst, exw, amax);
        edge_exp_sum<<<g1(N_EDGES * Hh, TB), TB, 0, stream>>>(dst, exw, amax, den);
        edge_aggregate<<<dim3(N_EDGES), 128, 0, stream>>>(v, src, dst, exw, den, agg);
    };
    auto zero_tail = [&](__bf16* base, int N) {   // zero pad rows [N_NODES, MPAD)
        long long n = (long long)(MPAD - N_NODES) * N;
        fill_bf16_zero<<<g1(n, TB), TB, 0, stream>>>(base + (long long)N_NODES * N, n);
    };

    // ---------------- layer 0 ----------------
    pack_qkv0<<<g1((long long)MPAD * 32, TB), TB, 0, stream>>>(obs, Aq0, Ak0, Av0);
    gemm(Aq0, wq0, 19, 32, HC, nullptr, fq, nullptr, 0);
    gemm(Ak0, wk0, 16, 32, HC, nullptr, fk, nullptr, 0);
    gemm(Av0, wv0, 18, 32, HC, nullptr, fv, nullptr, 0);
    edge_layer(fq, fk, fv);
    cvt_pad_bf16<<<g1((long long)MPAD * HC, TB), TB, 0, stream>>>(agg, bfA512, N_NODES, HC);
    zero_tail(bfA128, HID);
    gemm(bfA512, wo0, HC, HC, HID, bo0, nullptr, bfA128, 0);   // h0 (bf16, padded)

    // ---------------- layer 1 ----------------
    gemm(bfA128, wq1, HID, HID, HC, nullptr, fq, nullptr, 0);
    gemm(bfA128, wk1, HID, HID, HC, nullptr, fk, nullptr, 0);
    gemm(bfA128, wv1, HID, HID, HC, nullptr, fv, nullptr, 0);
    edge_layer(fq, fk, fv);
    cvt_pad_bf16<<<g1((long long)MPAD * HC, TB), TB, 0, stream>>>(agg, bfA512, N_NODES, HC);
    zero_tail(bfA128, HID);
    gemm(bfA512, wo1, HC, HC, HID, bo1, nullptr, bfA128, 0);   // h1 (bf16, padded)

    // ---------------- MLP head ----------------
    zero_tail(b256, 256);
    gemm(bfA128, w1, HID, HID, 256, b1, nullptr, b256, 1);     // relu(h1@w1+b1), bf16
    gemm(b256, w2, 256, 256, 256, b2, m2, nullptr, 1);         // relu(.@w2+b2), f32
    final_head<<<g1(N_NODES * 2, TB), TB, 0, stream>>>(m2, w3, b3, out);
}
